// AttentionalCopula_19971597926884
// MI455X (gfx1250) — matrix operand order repaired and checked by reading, not compiled, not run on hardware
//
#include <hip/hip_runtime.h>
#include <hip/hip_bf16.h>
#include <math.h>
#include <stdint.h>

// ---------------------------------------------------------------------------
// CDNA5 (gfx1250) wave32 WMMA types & fragment loaders
// ---------------------------------------------------------------------------
typedef __attribute__((ext_vector_type(16))) __bf16 v16bf;
typedef __attribute__((ext_vector_type(8)))  __bf16 v8bf;
typedef __attribute__((ext_vector_type(8)))  float  v8f;

// A-matrix (16x32 bf16) fragment from row-major [M][K] memory.
// ISA: lanes 0-15 row M=lane, K=[0..7]+[16..23]; lanes 16-31 K=[8..15]+[24..31].
__device__ __forceinline__ v16bf load_frag_a(const __bf16* base, int ld, int lane) {
  const __bf16* p = base + (size_t)(lane & 15) * (size_t)ld + ((lane >> 4) << 3);
  union { v16bf v; v8bf h[2]; } u;
  u.h[0] = *(const v8bf*)(p);
  u.h[1] = *(const v8bf*)(p + 16);
  return u.v;
}

// B-matrix (32x16 bf16) fragment from row-major Bt = [N][K] memory.
// Lane holds column N=lane&15, K=[0..7]+[16..23] (lanes 0-15) or [8..15]+[24..31] (lanes 16-31).
__device__ __forceinline__ v16bf load_frag_b(const __bf16* base, int ld, int lane) {
  const __bf16* p = base + (size_t)(lane & 15) * (size_t)ld + ((lane >> 4) << 3);
  union { v16bf v; v8bf h[2]; } u;
  u.h[0] = *(const v8bf*)(p);
  u.h[1] = *(const v8bf*)(p + 16);
  return u.v;
}

__device__ __forceinline__ v8f wmma_bf16(v16bf a, v16bf b, v8f c) {
  return __builtin_amdgcn_wmma_f32_16x16x32_bf16(false, a, false, b,
                                                 (short)0, c, false, false);
}

// Async copy of one 16-byte chunk global -> LDS (ASYNCcnt-tracked, CDNA5).
__device__ __forceinline__ void async_copy_b128(uint32_t lds_addr,
                                                const void* gptr) {
  asm volatile("global_load_async_to_lds_b128 %0, %1, off"
               :: "v"(lds_addr), "v"(gptr) : "memory");
}

// ---------------------------------------------------------------------------
// Direct-load batched GEMM (used for the small keys/vals projections, N=64):
// C[z] = A[z](MxK) * Bt[z]^T + bias[z]. Block tile 128x64, wave tile 32x32.
// ---------------------------------------------------------------------------
__global__ void __launch_bounds__(256)
gemm_bf16_wmma(const __bf16* __restrict__ A, size_t aStride, int lda,
               const __bf16* __restrict__ Bt, size_t bStride, int ldb,
               const float* __restrict__ bias, size_t biasStride,
               float* __restrict__ outF, __bf16* __restrict__ outB,
               size_t outStride, int ldc, int K, int relu)
{
  const int tid = threadIdx.x, lane = tid & 31, wave = tid >> 5;
  const int waveM = wave & 3, waveN = wave >> 2;
  const int z = blockIdx.z;

  const __bf16* Ab = A + (size_t)z * aStride +
                     (size_t)(blockIdx.x * 128 + waveM * 32) * (size_t)lda;
  const __bf16* Bb = Bt + (size_t)z * bStride +
                     (size_t)(blockIdx.y * 64 + waveN * 32) * (size_t)ldb;

  v8f acc[2][2] = {};
  for (int k0 = 0; k0 < K; k0 += 32) {
    v16bf a0 = load_frag_a(Ab + k0, lda, lane);
    v16bf a1 = load_frag_a(Ab + (size_t)16 * lda + k0, lda, lane);
    v16bf b0 = load_frag_b(Bb + k0, ldb, lane);
    v16bf b1 = load_frag_b(Bb + (size_t)16 * ldb + k0, ldb, lane);
    acc[0][0] = wmma_bf16(a0, b0, acc[0][0]);
    acc[0][1] = wmma_bf16(a0, b1, acc[0][1]);
    acc[1][0] = wmma_bf16(a1, b0, acc[1][0]);
    acc[1][1] = wmma_bf16(a1, b1, acc[1][1]);
  }

  const int nl = lane & 15, m0 = (lane >> 4) * 8;
  for (int mi = 0; mi < 2; ++mi)
    for (int ni = 0; ni < 2; ++ni) {
      const int gm = blockIdx.x * 128 + waveM * 32 + mi * 16 + m0;
      const int gn = blockIdx.y * 64 + waveN * 32 + ni * 16 + nl;
      const float bvv = bias ? bias[(size_t)z * biasStride + gn] : 0.0f;
      const size_t base = (size_t)z * outStride + (size_t)gm * ldc + gn;
      for (int i = 0; i < 8; ++i) {
        float vv = acc[mi][ni][i] + bvv;
        if (relu) vv = fmaxf(vv, 0.0f);
        const size_t o = base + (size_t)i * ldc;
        if (outB) outB[o] = (__bf16)vv;
        else      outF[o] = vv;
      }
    }
}

// ---------------------------------------------------------------------------
// Async-to-LDS double-buffered GEMM (the heavy path: att0, MLP, logits).
// Block tile 128(M) x 128(N) x 32(K). 8 waves, wave tile 32x64 (8 WMMA/step).
// Per K-stage: A tile 128x32 (8 KB) + B tile 128x32 (8 KB) = 1024 x 16B
// chunks, exactly 4 async_b128 per thread -> uniform s_wait_asynccnt 4 depth.
// LDS: 2 x 16 KB double buffer. M%128==0, N%128==0, K%32==0.
// ---------------------------------------------------------------------------
__global__ void __launch_bounds__(256)
gemm_bf16_wmma_async(const __bf16* __restrict__ A, int lda,
                     const __bf16* __restrict__ Bt, int ldb,
                     const float* __restrict__ bias,
                     float* __restrict__ outF, __bf16* __restrict__ outB,
                     int ldc, int K, int relu)
{
  __shared__ __align__(16) char smem[32768];   // [2][A 8KB | B 8KB]
  const int t = threadIdx.x, lane = t & 31, wave = t >> 5;
  const int waveM = wave & 3, waveN = wave >> 2;

  const __bf16* Ablk = A  + (size_t)(blockIdx.x * 128) * (size_t)lda;
  const __bf16* Bblk = Bt + (size_t)(blockIdx.y * 128) * (size_t)ldb;

  // Per-thread chunk assignment: chunks c and c+256 of each 512-chunk tile.
  const int c0 = t, c1 = t + 256;
  const __bf16* gA0 = Ablk + (size_t)(c0 >> 2) * lda + (c0 & 3) * 8;
  const __bf16* gA1 = Ablk + (size_t)(c1 >> 2) * lda + (c1 & 3) * 8;
  const __bf16* gB0 = Bblk + (size_t)(c0 >> 2) * ldb + (c0 & 3) * 8;
  const __bf16* gB1 = Bblk + (size_t)(c1 >> 2) * ldb + (c1 & 3) * 8;
  const uint32_t ldsBase = (uint32_t)(uintptr_t)(void*)smem;
  const uint32_t lA0 = ldsBase + (uint32_t)c0 * 16;
  const uint32_t lA1 = ldsBase + (uint32_t)c1 * 16;
  const uint32_t lB0 = ldsBase + 8192u + (uint32_t)c0 * 16;
  const uint32_t lB1 = ldsBase + 8192u + (uint32_t)c1 * 16;

  auto issue_stage = [&](int buf, int ks) {
    const uint32_t bo = (uint32_t)buf * 16384u;
    const size_t ke = (size_t)ks * 32;          // K offset in elements
    async_copy_b128(lA0 + bo, gA0 + ke);
    async_copy_b128(lA1 + bo, gA1 + ke);
    async_copy_b128(lB0 + bo, gB0 + ke);
    async_copy_b128(lB1 + bo, gB1 + ke);
  };

  const int nk = K >> 5;
  issue_stage(0, 0);

  v8f acc[2][4] = {};
  for (int ks = 0; ks < nk; ++ks) {
    if (ks + 1 < nk) {
      issue_stage((ks + 1) & 1, ks + 1);                 // prefetch next stage
      asm volatile("s_wait_asynccnt 4" ::: "memory");    // current stage done
    } else {
      asm volatile("s_wait_asynccnt 0" ::: "memory");
    }
    __syncthreads();                                      // tile visible to WG

    const __bf16* tA = (const __bf16*)(smem + (ks & 1) * 16384);
    const __bf16* tB = (const __bf16*)(smem + (ks & 1) * 16384 + 8192);
    v16bf a0 = load_frag_a(tA + (waveM * 32 + 0)  * 32, 32, lane);
    v16bf a1 = load_frag_a(tA + (waveM * 32 + 16) * 32, 32, lane);
#pragma unroll
    for (int ni = 0; ni < 4; ++ni) {
      v16bf bf = load_frag_b(tB + (waveN * 64 + ni * 16) * 32, 32, lane);
      acc[0][ni] = wmma_bf16(a0, bf, acc[0][ni]);
      acc[1][ni] = wmma_bf16(a1, bf, acc[1][ni]);
    }
    __syncthreads();   // protect buffer before it is overwritten next stage
  }

  const int nl = lane & 15, m0 = (lane >> 4) * 8;
  for (int mi = 0; mi < 2; ++mi)
    for (int ni = 0; ni < 4; ++ni) {
      const int gm = blockIdx.x * 128 + waveM * 32 + mi * 16 + m0;
      const int gn = blockIdx.y * 128 + waveN * 64 + ni * 16 + nl;
      const float bvv = bias ? bias[gn] : 0.0f;
      const size_t base = (size_t)gm * ldc + gn;
      for (int i = 0; i < 8; ++i) {
        float vv = acc[mi][ni][i] + bvv;
        if (relu) vv = fmaxf(vv, 0.0f);
        const size_t o = base + (size_t)i * ldc;
        if (outB) outB[o] = (__bf16)vv;
        else      outF[o] = vv;
      }
    }
}

// ---------------------------------------------------------------------------
// Fused attention for one (b, v, h): scores = q K^T * scale -> softmax -> P V
// ---------------------------------------------------------------------------
__global__ void __launch_bounds__(64)
attn_kernel(const __bf16* __restrict__ attB, const __bf16* __restrict__ keys,
            const __bf16* __restrict__ valsT, float* __restrict__ aout, int l)
{
  __shared__ float sS[32][64];
  __shared__ __align__(32) __bf16 sP[32][64];

  const int h = blockIdx.x, v = blockIdx.y, b = blockIdx.z;
  const int tid = threadIdx.x, lane = tid & 31, wave = tid >> 5;
  const size_t rowBase = ((size_t)(b * 64 + v)) * 32;

  const __bf16* Aq = attB + rowBase * 512 + h * 64;
  const __bf16* Kb = keys + ((size_t)((l * 8 + h) * 512 + b * 64)) * 64;

  v8f acc[2][2] = {};
  for (int k0 = 0; k0 < 64; k0 += 32) {
    v16bf a0 = load_frag_a(Aq + k0, 512, lane);
    v16bf a1 = load_frag_a(Aq + (size_t)16 * 512 + k0, 512, lane);
    v16bf b0 = load_frag_b(Kb + (size_t)(wave * 32) * 64 + k0, 64, lane);
    v16bf b1 = load_frag_b(Kb + (size_t)(wave * 32 + 16) * 64 + k0, 64, lane);
    acc[0][0] = wmma_bf16(a0, b0, acc[0][0]);
    acc[0][1] = wmma_bf16(a0, b1, acc[0][1]);
    acc[1][0] = wmma_bf16(a1, b0, acc[1][0]);
    acc[1][1] = wmma_bf16(a1, b1, acc[1][1]);
  }
  const float scale = 0.125f;
  const int nl = lane & 15, m0 = (lane >> 4) * 8;
  for (int mi = 0; mi < 2; ++mi)
    for (int ni = 0; ni < 2; ++ni)
      for (int i = 0; i < 8; ++i)
        sS[mi * 16 + m0 + i][wave * 32 + ni * 16 + nl] = acc[mi][ni][i] * scale;
  __syncthreads();

  if (tid < 32) {
    float mx = -1e30f;
    for (int c = 0; c < 64; ++c) mx = fmaxf(mx, sS[tid][c]);
    float ssum = 0.0f;
    for (int c = 0; c < 64; ++c) {
      float e = __expf(sS[tid][c] - mx);
      sS[tid][c] = e; ssum += e;
    }
    const float inv = 1.0f / ssum;
    for (int c = 0; c < 64; ++c) sP[tid][c] = (__bf16)(sS[tid][c] * inv);
  }
  __syncthreads();

  const __bf16* Vt = valsT + ((size_t)((l * 8 + h) * 8 + b)) * 4096;
  const __bf16* Pp = &sP[0][0];
  v8f acc2[2][2] = {};
  for (int k0 = 0; k0 < 64; k0 += 32) {
    v16bf a0 = load_frag_a(Pp + k0, 64, lane);
    v16bf a1 = load_frag_a(Pp + 16 * 64 + k0, 64, lane);
    v16bf b0 = load_frag_b(Vt + (size_t)(wave * 32) * 64 + k0, 64, lane);
    v16bf b1 = load_frag_b(Vt + (size_t)(wave * 32 + 16) * 64 + k0, 64, lane);
    acc2[0][0] = wmma_bf16(a0, b0, acc2[0][0]);
    acc2[0][1] = wmma_bf16(a0, b1, acc2[0][1]);
    acc2[1][0] = wmma_bf16(a1, b0, acc2[1][0]);
    acc2[1][1] = wmma_bf16(a1, b1, acc2[1][1]);
  }
  for (int mi = 0; mi < 2; ++mi)
    for (int ni = 0; ni < 2; ++ni)
      for (int i = 0; i < 8; ++i) {
        const size_t row = rowBase + mi * 16 + m0 + i;
        const int d = wave * 32 + ni * 16 + nl;
        aout[row * 512 + h * 64 + d] = acc2[mi][ni][i];
      }
}

// ---------------------------------------------------------------------------
// att = LayerNorm(att + add); writes f32 master and bf16 GEMM input.
// ---------------------------------------------------------------------------
__global__ void __launch_bounds__(128)
resln_kernel(float* __restrict__ att, const float* __restrict__ add,
             const float* __restrict__ g, const float* __restrict__ beta,
             __bf16* __restrict__ attB)
{
  __shared__ float red[128];
  const int row = blockIdx.x, tid = threadIdx.x;
  float* pa = att + (size_t)row * 512;
  const float* pd = add + (size_t)row * 512;

  float x[4]; float s = 0.0f;
  for (int j = 0; j < 4; ++j) { int c = tid + j * 128; x[j] = pa[c] + pd[c]; s += x[j]; }
  red[tid] = s; __syncthreads();
  for (int st = 64; st > 0; st >>= 1) { if (tid < st) red[tid] += red[tid + st]; __syncthreads(); }
  const float mean = red[0] * (1.0f / 512.0f);
  __syncthreads();

  float s2 = 0.0f;
  for (int j = 0; j < 4; ++j) { float d = x[j] - mean; s2 += d * d; }
  red[tid] = s2; __syncthreads();
  for (int st = 64; st > 0; st >>= 1) { if (tid < st) red[tid] += red[tid + st]; __syncthreads(); }
  const float rstd = rsqrtf(red[0] * (1.0f / 512.0f) + 1e-5f);

  for (int j = 0; j < 4; ++j) {
    const int c = tid + j * 128;
    const float y = (x[j] - mean) * rstd * g[c] + beta[c];
    pa[c] = y;
    attB[(size_t)row * 512 + c] = (__bf16)y;
  }
}

// ---------------------------------------------------------------------------
// Helpers
// ---------------------------------------------------------------------------
__global__ void cast_kernel(const float* __restrict__ in,
                            __bf16* __restrict__ out, int n) {
  const int i = blockIdx.x * 256 + threadIdx.x;
  if (i < n) out[i] = (__bf16)in[i];
}

__global__ void __launch_bounds__(256)
transpose_f32_bf16(const float* __restrict__ in, __bf16* __restrict__ out,
                   int R, int C)
{
  __shared__ float tile[32][33];
  const int z = blockIdx.z;
  const float* src = in + (size_t)z * R * C;
  __bf16* dst = out + (size_t)z * R * C;
  const int bx = blockIdx.x * 32, by = blockIdx.y * 32;
  const int tx = threadIdx.x, ty = threadIdx.y;
  for (int j = ty; j < 32; j += 8)
    tile[j][tx] = src[(size_t)(by + j) * C + bx + tx];
  __syncthreads();
  for (int j = ty; j < 32; j += 8)
    dst[(size_t)(bx + j) * R + by + tx] = (__bf16)tile[tx][j];
}

__global__ void __launch_bounds__(256)
transpose_bf16_64(const __bf16* __restrict__ in, __bf16* __restrict__ out)
{
  __shared__ __bf16 t[64][65];
  const __bf16* src = in + (size_t)blockIdx.x * 4096;
  __bf16* dst = out + (size_t)blockIdx.x * 4096;
  const int tx = threadIdx.x, ty = threadIdx.y;
  for (int r = ty; r < 64; r += 4) t[r][tx] = src[r * 64 + tx];
  __syncthreads();
  for (int r = ty; r < 64; r += 4) dst[r * 64 + tx] = t[tx][r];
}

__global__ void bcast_kernel(const float* __restrict__ src,
                             float* __restrict__ attF, __bf16* __restrict__ attB)
{
  const size_t idx = (size_t)blockIdx.x * 256 + threadIdx.x;
  const int col = (int)(idx & 511);
  const size_t row = idx >> 9;
  const float vv = src[(row >> 5) * 512 + col];
  attF[idx] = vv;
  attB[idx] = (__bf16)vv;
}

__device__ __forceinline__ unsigned mixhash(unsigned x) {
  x ^= x >> 16; x *= 0x7feb352dU; x ^= x >> 15; x *= 0x846ca68bU; x ^= x >> 16;
  return x;
}
__global__ void __launch_bounds__(128)
sample_kernel(const float* __restrict__ logits, float* __restrict__ out)
{
  __shared__ float sv[128];
  __shared__ int si[128];
  const int row = blockIdx.x, c = threadIdx.x;
  const float lg = logits[(size_t)row * 128 + c];
  const unsigned h = mixhash(0x9E3779B9U ^ (unsigned)(row * 131 + c));
  const float u = ((float)h + 0.5f) * 2.3283064e-10f;
  sv[c] = lg - logf(-logf(u));
  si[c] = c;
  __syncthreads();
  for (int s = 64; s > 0; s >>= 1) {
    if (c < s && sv[c + s] > sv[c]) { sv[c] = sv[c + s]; si[c] = si[c + s]; }
    __syncthreads();
  }
  if (c == 0) {
    const unsigned h2 = mixhash(0xABCDEF01U ^ (unsigned)row);
    const float u2 = ((float)h2 + 0.5f) * 2.3283064e-10f;
    out[row] = ((float)si[0] + u2) * (1.0f / 128.0f);
  }
}

// ---------------------------------------------------------------------------
// Host orchestration
// ---------------------------------------------------------------------------
extern "C" void kernel_launch(void* const* d_in, const int* in_sizes, int n_in,
                              void* d_out, int out_size, void* d_ws, size_t ws_size,
                              hipStream_t stream) {
  (void)in_sizes; (void)n_in; (void)out_size; (void)ws_size;
  const float* flow = (const float*)d_in[0];
  const float* Wsh  = (const float*)d_in[1];
  const float* bsh  = (const float*)d_in[2];
  const float* Wk   = (const float*)d_in[3];
  const float* bk   = (const float*)d_in[4];
  const float* Wv   = (const float*)d_in[5];
  const float* bvp  = (const float*)d_in[6];
  const float* ln1g = (const float*)d_in[7];
  const float* ln1b = (const float*)d_in[8];
  const float* W1   = (const float*)d_in[9];
  const float* b1   = (const float*)d_in[10];
  const float* W2   = (const float*)d_in[11];
  const float* b2   = (const float*)d_in[12];
  const float* W3   = (const float*)d_in[13];
  const float* b3   = (const float*)d_in[14];
  const float* ln2g = (const float*)d_in[15];
  const float* ln2b = (const float*)d_in[16];
  const float* Wd   = (const float*)d_in[17];
  const float* bd   = (const float*)d_in[18];
  float* out = (float*)d_out;

  char* ws = (char*)d_ws;
  size_t off = 0;
  auto alloc = [&](size_t bytes) -> char* {
    off = (off + 255) & ~(size_t)255;
    char* p = ws + off; off += bytes; return p;
  };
  __bf16* flowB = (__bf16*)alloc((size_t)512 * 512 * 2);
  __bf16* WshT  = (__bf16*)alloc((size_t)512 * 512 * 2);
  __bf16* WkT   = (__bf16*)alloc((size_t)32 * 64 * 512 * 2);
  __bf16* WvT   = (__bf16*)alloc((size_t)32 * 64 * 512 * 2);
  __bf16* W1T   = (__bf16*)alloc((size_t)4 * 1024 * 512 * 2);
  __bf16* W2T   = (__bf16*)alloc((size_t)4 * 1024 * 1024 * 2);
  __bf16* W3T   = (__bf16*)alloc((size_t)4 * 512 * 1024 * 2);
  __bf16* WdT   = (__bf16*)alloc((size_t)128 * 512 * 2);
  __bf16* keysB = (__bf16*)alloc((size_t)32 * 512 * 64 * 2);
  __bf16* valsB = (__bf16*)alloc((size_t)32 * 512 * 64 * 2);
  __bf16* valsT = (__bf16*)alloc((size_t)32 * 512 * 64 * 2);
  float*  attF  = (float*) alloc((size_t)16384 * 512 * 4);
  __bf16* attB  = (__bf16*)alloc((size_t)16384 * 512 * 2);
  float*  tmpF  = (float*) alloc((size_t)16384 * 512 * 4);
  __bf16* h1    = (__bf16*)alloc((size_t)16384 * 1024 * 2);
  __bf16* h2    = (__bf16*)alloc((size_t)16384 * 1024 * 2);
  float*  logit = (float*) alloc((size_t)16384 * 128 * 4);

  // --- prep: casts + weight transposes (Bt = [N][K] bf16) ---
  cast_kernel<<<1024, 256, 0, stream>>>(flow, flowB, 512 * 512);
  transpose_f32_bf16<<<dim3(16, 16, 1),  dim3(32, 8), 0, stream>>>(Wsh, WshT, 512, 512);
  transpose_f32_bf16<<<dim3(2, 16, 32),  dim3(32, 8), 0, stream>>>(Wk, WkT, 512, 64);
  transpose_f32_bf16<<<dim3(2, 16, 32),  dim3(32, 8), 0, stream>>>(Wv, WvT, 512, 64);
  transpose_f32_bf16<<<dim3(32, 16, 4),  dim3(32, 8), 0, stream>>>(W1, W1T, 512, 1024);
  transpose_f32_bf16<<<dim3(32, 32, 4),  dim3(32, 8), 0, stream>>>(W2, W2T, 1024, 1024);
  transpose_f32_bf16<<<dim3(16, 32, 4),  dim3(32, 8), 0, stream>>>(W3, W3T, 1024, 512);
  transpose_f32_bf16<<<dim3(4, 16, 1),   dim3(32, 8), 0, stream>>>(Wd, WdT, 512, 128);

  // --- keys / vals: batched over z = l*H+h (direct GEMM, N=64) ---
  gemm_bf16_wmma<<<dim3(4, 1, 32), 256, 0, stream>>>(
      flowB, 0, 512, WkT, (size_t)64 * 512, 512, bk, 64,
      nullptr, keysB, (size_t)512 * 64, 64, 512, 0);
  gemm_bf16_wmma<<<dim3(4, 1, 32), 256, 0, stream>>>(
      flowB, 0, 512, WvT, (size_t)64 * 512, 512, bvp, 64,
      nullptr, valsB, (size_t)512 * 64, 64, 512, 0);
  transpose_bf16_64<<<256, dim3(64, 4), 0, stream>>>(valsB, valsT);

  // --- att0 = flow @ W_shift + b_shift (async-LDS GEMM), broadcast ---
  gemm_bf16_wmma_async<<<dim3(4, 4, 1), 256, 0, stream>>>(
      flowB, 512, WshT, 512, bsh, tmpF, nullptr, 512, 512, 0);
  bcast_kernel<<<32768, 256, 0, stream>>>(tmpF, attF, attB);

  // --- 4 transformer-copula layers ---
  for (int l = 0; l < 4; ++l) {
    attn_kernel<<<dim3(8, 64, 8), 64, 0, stream>>>(attB, keysB, valsT, tmpF, l);
    resln_kernel<<<16384, 128, 0, stream>>>(attF, tmpF, ln1g + l * 512, ln1b + l * 512, attB);
    gemm_bf16_wmma_async<<<dim3(128, 8, 1), 256, 0, stream>>>(
        attB, 512, W1T + (size_t)l * 1024 * 512, 512, b1 + l * 1024,
        nullptr, h1, 1024, 512, 1);
    gemm_bf16_wmma_async<<<dim3(128, 8, 1), 256, 0, stream>>>(
        h1, 1024, W2T + (size_t)l * 1024 * 1024, 1024, b2 + l * 1024,
        nullptr, h2, 1024, 1024, 1);
    gemm_bf16_wmma_async<<<dim3(128, 4, 1), 256, 0, stream>>>(
        h2, 1024, W3T + (size_t)l * 512 * 1024, 1024, b3 + l * 512,
        tmpF, nullptr, 512, 1024, 0);
    resln_kernel<<<16384, 128, 0, stream>>>(attF, tmpF, ln2g + l * 512, ln2b + l * 512, attB);
  }

  // --- distribution head + categorical sampling ---
  gemm_bf16_wmma_async<<<dim3(128, 1, 1), 256, 0, stream>>>(
      attB, 512, WdT, 512, bd, logit, nullptr, 128, 512, 0);
  sample_kernel<<<16384, 128, 0, stream>>>(logit, out);
}
